// Kernel_11041417490697712865_53472342835844
// MI455X (gfx1250) — compile-verified, matrix operand order checked by reading, not run on hardware
//
#include <hip/hip_runtime.h>
#include <hip/hip_bf16.h>
#include <stdint.h>

// ---------------------------------------------------------------------------
// MI455X (gfx1250) pipeline:
//   prep34 : t3 (depthwise 1x5 dil2) + t4 (grouped 3x1) -> bf16 [n][c][hw]
//   prep6  : roll(+2,ch) + height unfold {-3,0,+3}      -> bf16 [n][p][768]
//   cvtw7  : w7 f32 -> bf16 row-major [o][768]
//   gemm_t7: D = W7A (256x768) x T6T^T  -> store transposed as T7T [n][p][256]
//   gemm_s : D = T3 (256x4096) x T4^T / 64 -> S = t8^T  [n][d][256]
//   gemm_t9: D = S  (256x256)  x T7T^T / 16 -> f32 out  [n][d][hw]
// GEMMs: 32x32 output tile per wave32 (2x2 register blocking) via
// v_wmma_f32_16x16x32_bf16; 4 fragment loads (128B) feed 4 WMMAs (64 KFLOP)
// per K-step -> ~512 FLOP/byte, halving L2 traffic vs 16x16-per-wave.
// ---------------------------------------------------------------------------

typedef __bf16 bf16_t;
typedef __attribute__((ext_vector_type(16))) __bf16 v16bf;
typedef __attribute__((ext_vector_type(8)))  __bf16 v8bf;
typedef __attribute__((ext_vector_type(8)))  float  v8f;

#define NB 16
#define C  256
#define H  64
#define W  64
#define HW 4096
#define K7 768   // 3*C

// Per-lane bf16 fragment load, 16x32 tile, row-major along K (ISA 7.12.2):
// lane 0-15: row M=lane, kb=0; lane 16-31: row M=lane-16, kb=8.
__device__ __forceinline__ v16bf ldfrag(const bf16_t* p) {
    v8bf lo = *(const v8bf*)(p);
    v8bf hi = *(const v8bf*)(p + 16);
    v16bf r;
#pragma unroll
    for (int i = 0; i < 8; ++i) { r[i] = lo[i]; r[i + 8] = hi[i]; }
    return r;
}

struct Acc2x2 { v8f a00, a01, a10, a11; };

// 32x32 output tile per wave: 2 M-subtiles x 2 N-subtiles, f32 accumulate.
__device__ __forceinline__ Acc2x2 wave_gemm_2x2(const bf16_t* A, const bf16_t* Bt,
                                                int tm, int tn, int K) {
    const int lane = threadIdx.x & 31;
    const int lm   = lane & 15;
    const int kb   = (lane >> 4) << 3;
    const bf16_t* ap = A  + (size_t)(tm + lm) * K + kb;   // +16*K for subtile 1
    const bf16_t* bp = Bt + (size_t)(tn + lm) * K + kb;
    const size_t  sub = (size_t)16 * K;

    Acc2x2 acc;
#pragma unroll
    for (int i = 0; i < 8; ++i) {
        acc.a00[i] = 0.f; acc.a01[i] = 0.f; acc.a10[i] = 0.f; acc.a11[i] = 0.f;
    }
    for (int k0 = 0; k0 < K; k0 += 32) {
        // speculative stream prefetch (global_prefetch_b8)
        __builtin_prefetch(ap + k0 + 128, 0, 1);
        __builtin_prefetch(ap + sub + k0 + 128, 0, 1);
        __builtin_prefetch(bp + k0 + 128, 0, 1);
        __builtin_prefetch(bp + sub + k0 + 128, 0, 1);
        v16bf a0 = ldfrag(ap + k0);
        v16bf a1 = ldfrag(ap + sub + k0);
        v16bf b0 = ldfrag(bp + k0);
        v16bf b1 = ldfrag(bp + sub + k0);
        acc.a00 = __builtin_amdgcn_wmma_f32_16x16x32_bf16(false, a0, false, b0,
                                                          (short)0, acc.a00, false, false);
        acc.a01 = __builtin_amdgcn_wmma_f32_16x16x32_bf16(false, a0, false, b1,
                                                          (short)0, acc.a01, false, false);
        acc.a10 = __builtin_amdgcn_wmma_f32_16x16x32_bf16(false, a1, false, b0,
                                                          (short)0, acc.a10, false, false);
        acc.a11 = __builtin_amdgcn_wmma_f32_16x16x32_bf16(false, a1, false, b1,
                                                          (short)0, acc.a11, false, false);
    }
    return acc;
}

// ---------------- prep kernels ----------------

__global__ __launch_bounds__(256)
void prep34_kernel(const float* __restrict__ X, const float* __restrict__ w3,
                   const float* __restrict__ w4,
                   bf16_t* __restrict__ T3, bf16_t* __restrict__ T4) {
    int t = blockIdx.x * 256 + threadIdx.x;           // n*c*hw threads
    int xx = t & 63, yy = (t >> 6) & 63, ch = (t >> 12) & 255, nb = t >> 20;

    // t3: depthwise width conv, taps at x + 2k - 4
    const float* xrow = X + ((size_t)(nb * C + ch)) * HW + yy * W;
    float s3 = 0.f;
#pragma unroll
    for (int k = 0; k < 5; ++k) {
        int xs = xx + 2 * k - 4;
        if (xs >= 0 && xs < W) s3 += xrow[xs] * w3[ch * 5 + k];
    }

    // t4: grouped conv, 8 in-channels x 3 height taps
    int icb = (ch >> 3) << 3;
    const float* wk = w4 + ch * 24;
    float s4 = 0.f;
#pragma unroll
    for (int i = 0; i < 8; ++i) {
        const float* xc = X + ((size_t)(nb * C + icb + i)) * HW + xx;
#pragma unroll
        for (int dy = 0; dy < 3; ++dy) {
            int ys = yy + dy - 1;
            if (ys >= 0 && ys < H) s4 += xc[ys * W] * wk[i * 3 + dy];
        }
    }
    T3[t] = (bf16_t)s3;
    T4[t] = (bf16_t)s4;
}

__global__ __launch_bounds__(256)
void prep6_kernel(const float* __restrict__ X, bf16_t* __restrict__ T6T) {
    int t = blockIdx.x * 256 + threadIdx.x;           // n*c*hw threads
    int p = t & 4095, ch = (t >> 12) & 255, nb = t >> 20;
    int yy = p >> 6, xx = p & 63;
    int sc = (ch + C - 2) & 255;                      // roll(+2) source channel
    const float* xc = X + ((size_t)(nb * C + sc)) * HW;
    size_t ob = ((size_t)nb * HW + p) * K7 + ch * 3;  // [n][p][k], k = ch*3 + j
#pragma unroll
    for (int j = 0; j < 3; ++j) {
        int ys = yy + 3 * j - 3;
        float v = (ys >= 0 && ys < H) ? xc[ys * W + xx] : 0.f;
        T6T[ob + j] = (bf16_t)v;
    }
}

__global__ __launch_bounds__(256)
void cvtw7_kernel(const float* __restrict__ w7, bf16_t* __restrict__ W7A) {
    int t = blockIdx.x * 256 + threadIdx.x;           // c * 3c threads
    W7A[t] = (bf16_t)w7[t];
}

// ---------------- GEMM kernels ----------------

// t7: A = W7A (256 x 768), B^T = T6T[n] (4096 x 768); store D transposed bf16.
__global__ __launch_bounds__(256)
void gemm_t7_kernel(const bf16_t* __restrict__ W7A, const bf16_t* __restrict__ T6T,
                    bf16_t* __restrict__ T7T) {
    int wid = (blockIdx.x * 256 + threadIdx.x) >> 5;
    int nt = wid & 127, mt = (wid >> 7) & 7, nb = wid >> 10;   // 32-wide tiles
    const bf16_t* Bt = T6T + (size_t)nb * HW * K7;
    Acc2x2 acc = wave_gemm_2x2(W7A, Bt, mt * 32, nt * 32, K7);

    const int lane = threadIdx.x & 31;
    const int lm = lane & 15, hi = lane >> 4;
    const v8f* av[4] = { &acc.a00, &acc.a01, &acc.a10, &acc.a11 };
#pragma unroll
    for (int s = 0; s < 4; ++s) {
        int mo = (s >> 1) * 16, no = (s & 1) * 16;
#pragma unroll
        for (int r = 0; r < 8; ++r) {
            int o = mt * 32 + mo + r + 8 * hi;        // M (output channel)
            int p = nt * 32 + no + lm;                // N (pixel)
            T7T[((size_t)nb * HW + p) * C + o] = (bf16_t)((*av[s])[r]);
        }
    }
}

// S = t8^T: A = T3[n] (256 x 4096), B^T = T4[n] (256 x 4096); scale 1/64.
__global__ __launch_bounds__(256)
void gemm_s_kernel(const bf16_t* __restrict__ T3, const bf16_t* __restrict__ T4,
                   bf16_t* __restrict__ S) {
    int wid = (blockIdx.x * 256 + threadIdx.x) >> 5;
    int nt = wid & 7, mt = (wid >> 3) & 7, nb = wid >> 6;      // 32-wide tiles
    const bf16_t* A  = T3 + (size_t)nb * C * HW;
    const bf16_t* Bt = T4 + (size_t)nb * C * HW;
    Acc2x2 acc = wave_gemm_2x2(A, Bt, mt * 32, nt * 32, HW);

    const int lane = threadIdx.x & 31;
    const int lm = lane & 15, hi = lane >> 4;
    const v8f* av[4] = { &acc.a00, &acc.a01, &acc.a10, &acc.a11 };
#pragma unroll
    for (int s = 0; s < 4; ++s) {
        int mo = (s >> 1) * 16, no = (s & 1) * 16;
#pragma unroll
        for (int r = 0; r < 8; ++r) {
            int d  = mt * 32 + mo + r + 8 * hi;       // M (t3 channel)
            int cc = nt * 32 + no + lm;               // N (t4 channel)
            S[(size_t)nb * C * C + (size_t)d * C + cc] =
                (bf16_t)((*av[s])[r] * 0.015625f);
        }
    }
}

// t9: A = S[n] (256 x 256), B^T = T7T[n] (4096 x 256); scale 1/16, f32 out.
__global__ __launch_bounds__(256)
void gemm_t9_kernel(const bf16_t* __restrict__ S, const bf16_t* __restrict__ T7T,
                    float* __restrict__ out) {
    int wid = (blockIdx.x * 256 + threadIdx.x) >> 5;
    int nt = wid & 127, mt = (wid >> 7) & 7, nb = wid >> 10;   // 32-wide tiles
    const bf16_t* A  = S   + (size_t)nb * C * C;
    const bf16_t* Bt = T7T + (size_t)nb * HW * C;
    Acc2x2 acc = wave_gemm_2x2(A, Bt, mt * 32, nt * 32, C);

    const int lane = threadIdx.x & 31;
    const int lm = lane & 15, hi = lane >> 4;
    const v8f* av[4] = { &acc.a00, &acc.a01, &acc.a10, &acc.a11 };
#pragma unroll
    for (int s = 0; s < 4; ++s) {
        int mo = (s >> 1) * 16, no = (s & 1) * 16;
#pragma unroll
        for (int r = 0; r < 8; ++r) {
            int d = mt * 32 + mo + r + 8 * hi;        // M (output channel)
            int p = nt * 32 + no + lm;                // N (pixel)
            out[((size_t)nb * C + d) * HW + p] = (*av[s])[r] * 0.0625f;
        }
    }
}

// ---------------- launcher ----------------

extern "C" void kernel_launch(void* const* d_in, const int* in_sizes, int n_in,
                              void* d_out, int out_size, void* d_ws, size_t ws_size,
                              hipStream_t stream) {
    const float* x  = (const float*)d_in[0];
    const float* w3 = (const float*)d_in[1];
    const float* w4 = (const float*)d_in[2];
    const float* w7 = (const float*)d_in[3];
    float* out = (float*)d_out;

    uint8_t* ws = (uint8_t*)d_ws;
    bf16_t* T3  = (bf16_t*)ws;  ws += (size_t)NB * C * HW * 2;      //  32 MB
    bf16_t* T4  = (bf16_t*)ws;  ws += (size_t)NB * C * HW * 2;      //  32 MB
    bf16_t* T6T = (bf16_t*)ws;  ws += (size_t)NB * HW * K7 * 2;     //  96 MB
    bf16_t* W7A = (bf16_t*)ws;  ws += (size_t)C * K7 * 2;           // 0.4 MB
    bf16_t* T7T = (bf16_t*)ws;  ws += (size_t)NB * HW * C * 2;      //  32 MB
    bf16_t* S   = (bf16_t*)ws;  ws += (size_t)NB * C * C * 2;       //   2 MB

    prep34_kernel<<<(NB * C * HW) / 256, 256, 0, stream>>>(x, w3, w4, T3, T4);
    prep6_kernel <<<(NB * C * HW) / 256, 256, 0, stream>>>(x, T6T);
    cvtw7_kernel <<<(C * K7)      / 256, 256, 0, stream>>>(w7, W7A);

    // waves = NB * (M/32) * (N/32); blockDim 256 = 8 wave32 tiles per block
    gemm_t7_kernel<<<(NB * 8 * 128 * 32) / 256, 256, 0, stream>>>(W7A, T6T, T7T);
    gemm_s_kernel <<<(NB * 8 * 8   * 32) / 256, 256, 0, stream>>>(T3, T4, S);
    gemm_t9_kernel<<<(NB * 8 * 128 * 32) / 256, 256, 0, stream>>>(S, T7T, out);
}
